// SimpleGNN_5712306504439
// MI455X (gfx1250) — matrix operand order repaired and checked by reading, not compiled, not run on hardware
//
#include <hip/hip_runtime.h>
#include <hip/hip_bf16.h>
#include <stdint.h>

// ---------------------------------------------------------------------------
// GNN pipeline for gfx1250 (MI455X): SAGE -> LN/ELU -> GAT -> LN/ELU -> SAGE
// -> MLP.  Dense per-node GEMMs run on v_wmma_f32_16x16x32_f16 (wave32 WMMA)
// with B fragments staged block-wide into LDS via the CDNA5 async-LDS DMA
// path (global_load_async_to_lds_b128 / s_wait_asynccnt).  Graph aggregation
// runs as wave-per-edge float4 gathers + f32 atomics.
// ---------------------------------------------------------------------------

typedef __attribute__((ext_vector_type(16))) _Float16 v16h;
typedef __attribute__((ext_vector_type(8)))  float    v8f;

#define HEADS 4
#define DH 128

// ------------------------------ utility ------------------------------------

__global__ void zero_kernel(float* __restrict__ p, long n) {
    long i = (long)blockIdx.x * blockDim.x + threadIdx.x;
    if (i < n) p[i] = 0.0f;
}

// Pack fp32 row-major weight W[K][NOUT] into f16 B-fragment layout:
// packed[(((kb*NBtot + nbg)*32 + lane)*16 + e)] = W[kb*32 + (lane>>4)*16 + e][nbg*16 + (lane&15)]
__global__ void pack_w_kernel(const float* __restrict__ W, _Float16* __restrict__ out,
                              int K, int NOUT) {
    int p = blockIdx.x * blockDim.x + threadIdx.x;
    int total = K * NOUT;
    if (p >= total) return;
    int e    = p & 15;
    int lane = (p >> 4) & 31;
    int NBtot = NOUT >> 4;
    int nbg  = (p >> 9) % NBtot;
    int kb   = (p >> 9) / NBtot;
    int kk = kb * 32 + (lane >> 4) * 16 + e;
    int n  = nbg * 16 + (lane & 15);
    out[p] = (_Float16)W[(size_t)kk * NOUT + n];
}

// --------------------------- WMMA GEMM -------------------------------------
// out[M x NOUT](block cols = NB*16 at blockIdx.y) = A1@W1 (+ A2@W2 if DUAL)
// + bias, optional ReLU.  256 threads = 8 waves; wave w computes rows
// [blockIdx.x*128 + w*16, +16).  K = KB*32 (compile time, fully unrolled).
// B fragments are staged once per block into LDS with async-LDS DMA; all 8
// waves share them via ds_load_b128.
template<int NB, int KB, bool DUAL>
__global__ __launch_bounds__(256) void gemm_kernel(
    const float* __restrict__ A1, const v16h* __restrict__ W1,
    const float* __restrict__ A2, const v16h* __restrict__ W2,
    const float* __restrict__ bias, float* __restrict__ out,
    int M, int NOUT, int act)
{
    constexpr int NPASS = DUAL ? 2 : 1;
    constexpr int K = KB * 32;
    __shared__ _Float16 sB[NPASS * KB * NB * 32 * 16];

    const int tid  = threadIdx.x;
    const int lane = tid & 31;
    const int wave = tid >> 5;
    const int g    = lane >> 4;      // lane group (0..1)
    const int mr   = lane & 15;      // A row / C col within tile
    const int tile_m = blockIdx.x * 128 + wave * 16;
    const int col0   = blockIdx.y * (NB * 16);
    const int NBtot  = NOUT >> 4;

    // ---- stage packed B fragments to LDS via CDNA5 async DMA ----
    // generic-ptr low 32 bits == LDS byte address
    const unsigned sbase = (unsigned)(uintptr_t)(&sB[0]);
#pragma unroll
    for (int s = 0; s < NPASS; ++s) {
        const v16h* Wp = (DUAL && s) ? W2 : W1;
#pragma unroll
        for (int kb = 0; kb < KB; ++kb) {
            const uint64_t gbase =
                (uint64_t)(const char*)(Wp + ((size_t)kb * NBtot + blockIdx.y * NB) * 32);
            const unsigned lbase = sbase + (unsigned)(((s * KB + kb) * NB * 32) * 32);
            constexpr int CHUNKS = NB * 64;          // 16-byte chunks per kb piece
            for (int c = tid; c < CHUNKS; c += 256) {
                unsigned laddr = lbase + c * 16;
                uint64_t gaddr = gbase + (uint64_t)c * 16;
                asm volatile("global_load_async_to_lds_b128 %0, %1, off"
                             :: "v"(laddr), "v"(gaddr) : "memory");
            }
        }
    }
    asm volatile("s_wait_asynccnt 0x0" ::: "memory");
    __syncthreads();

    v8f acc[NB];
#pragma unroll
    for (int i = 0; i < NB; ++i) acc[i] = (v8f){0.f,0.f,0.f,0.f,0.f,0.f,0.f,0.f};

    int m = tile_m + mr;
    if (m >= M) m = M - 1;                   // clamp: WMMA needs full EXEC

#pragma unroll
    for (int s = 0; s < NPASS; ++s) {
        const float* arow = ((DUAL && s) ? A2 : A1) + (size_t)m * K;
        v16h a[KB];
#pragma unroll
        for (int kb = 0; kb < KB; ++kb) {
            const float* p = arow + kb * 32 + g * 16;
            float4 f0 = *(const float4*)(p);
            float4 f1 = *(const float4*)(p + 4);
            float4 f2 = *(const float4*)(p + 8);
            float4 f3 = *(const float4*)(p + 12);
            v16h& a_ = a[kb];
            a_[0]=(_Float16)f0.x;  a_[1]=(_Float16)f0.y;  a_[2]=(_Float16)f0.z;  a_[3]=(_Float16)f0.w;
            a_[4]=(_Float16)f1.x;  a_[5]=(_Float16)f1.y;  a_[6]=(_Float16)f1.z;  a_[7]=(_Float16)f1.w;
            a_[8]=(_Float16)f2.x;  a_[9]=(_Float16)f2.y;  a_[10]=(_Float16)f2.z; a_[11]=(_Float16)f2.w;
            a_[12]=(_Float16)f3.x; a_[13]=(_Float16)f3.y; a_[14]=(_Float16)f3.z; a_[15]=(_Float16)f3.w;
        }
#pragma unroll
        for (int nb = 0; nb < NB; ++nb) {
#pragma unroll
            for (int kb = 0; kb < KB; ++kb) {
                const v16h b =
                    *(const v16h*)&sB[(((s * KB + kb) * NB + nb) * 32 + lane) * 16];
                acc[nb] = __builtin_amdgcn_wmma_f32_16x16x32_f16(
                    false, a[kb], false, b, (short)0, acc[nb], false, false);
            }
        }
    }

    // epilogue: C layout -> lane col = lane&15, VGPR r -> row g*8 + r
#pragma unroll
    for (int nb = 0; nb < NB; ++nb) {
        int col = col0 + nb * 16 + mr;
        float bv = bias ? bias[col] : 0.0f;
#pragma unroll
        for (int r = 0; r < 8; ++r) {
            int row = tile_m + g * 8 + r;
            if (row < M) {
                float v = acc[nb][r] + bv;
                if (act == 1) v = v > 0.f ? v : 0.f;
                out[(size_t)row * NOUT + col] = v;
            }
        }
    }
}

// ------------------------ LayerNorm + optional ELU -------------------------
__global__ __launch_bounds__(256) void ln_act_kernel(
    float* __restrict__ h, const float* __restrict__ gamma,
    const float* __restrict__ beta, int M, int act)
{
    int wave = threadIdx.x >> 5, lane = threadIdx.x & 31;
    int row = blockIdx.x * 8 + wave;
    if (row >= M) return;
    float4 v = ((const float4*)(h + (size_t)row * DH))[lane];
    float s  = v.x + v.y + v.z + v.w;
    float ss = v.x*v.x + v.y*v.y + v.z*v.z + v.w*v.w;
    for (int o = 16; o > 0; o >>= 1) {
        s  += __shfl_xor(s, o, 32);
        ss += __shfl_xor(ss, o, 32);
    }
    float mu  = s * (1.0f / DH);
    float var = ss * (1.0f / DH) - mu * mu;
    float rs  = rsqrtf(var + 1e-5f);
    float4 gv = ((const float4*)gamma)[lane];
    float4 bv = ((const float4*)beta)[lane];
    float4 o4;
    o4.x = (v.x - mu) * rs * gv.x + bv.x;
    o4.y = (v.y - mu) * rs * gv.y + bv.y;
    o4.z = (v.z - mu) * rs * gv.z + bv.z;
    o4.w = (v.w - mu) * rs * gv.w + bv.w;
    if (act) {
        o4.x = o4.x > 0.f ? o4.x : expm1f(o4.x);
        o4.y = o4.y > 0.f ? o4.y : expm1f(o4.y);
        o4.z = o4.z > 0.f ? o4.z : expm1f(o4.z);
        o4.w = o4.w > 0.f ? o4.w : expm1f(o4.w);
    }
    ((float4*)(h + (size_t)row * DH))[lane] = o4;
}

// --------------------------- graph aggregation -----------------------------
__global__ void deg_kernel(const int* __restrict__ dst, float* __restrict__ deg, int E) {
    int e = blockIdx.x * blockDim.x + threadIdx.x;
    if (e < E) atomicAdd(&deg[dst[e]], 1.0f);
}

// msg_sum[dst] += feat[src] ; one wave per edge, float4 per lane
__global__ __launch_bounds__(256) void scatter_feat_kernel(
    const float* __restrict__ feat, const int* __restrict__ src,
    const int* __restrict__ dst, float* __restrict__ out, int E)
{
    int wave = threadIdx.x >> 5, lane = threadIdx.x & 31;
    int e = blockIdx.x * 8 + wave;
    if (e >= E) return;
    int s = src[e], d = dst[e];
    float4 v = ((const float4*)(feat + (size_t)s * DH))[lane];
    float* o = out + (size_t)d * DH + lane * 4;
    atomicAdd(o + 0, v.x);
    atomicAdd(o + 1, v.y);
    atomicAdd(o + 2, v.z);
    atomicAdd(o + 3, v.w);
}

__global__ void div_deg_kernel(float* __restrict__ agg, const float* __restrict__ deg, long n) {
    long i = (long)blockIdx.x * blockDim.x + threadIdx.x;
    if (i < n) agg[i] /= fmaxf(deg[i >> 7], 1.0f);
}

// ------------------------------- GAT ---------------------------------------
__global__ __launch_bounds__(256) void attn_coef_kernel(
    const float* __restrict__ xw, const float* __restrict__ att_src,
    const float* __restrict__ att_dst, float* __restrict__ a_s,
    float* __restrict__ a_d, int N)
{
    int wave = threadIdx.x >> 5, lane = threadIdx.x & 31;
    int idx = blockIdx.x * 8 + wave;
    if (idx >= N * HEADS) return;
    int n = idx >> 2, h = idx & 3;
    float4 x  = ((const float4*)(xw + (size_t)n * (HEADS * DH) + h * DH))[lane];
    float4 as = ((const float4*)(att_src + h * DH))[lane];
    float4 ad = ((const float4*)(att_dst + h * DH))[lane];
    float s1 = x.x*as.x + x.y*as.y + x.z*as.z + x.w*as.w;
    float s2 = x.x*ad.x + x.y*ad.y + x.z*ad.z + x.w*ad.w;
    for (int o = 16; o > 0; o >>= 1) {
        s1 += __shfl_xor(s1, o, 32);
        s2 += __shfl_xor(s2, o, 32);
    }
    if (lane == 0) { a_s[idx] = s1; a_d[idx] = s2; }
}

__device__ inline void edge_sd(const int* src, const int* dst, int E, int e2,
                               int& s, int& d) {
    if (e2 < E) { s = src[e2]; d = dst[e2]; }
    else        { s = d = e2 - E; }            // self-loop
}

__device__ inline unsigned f32_key(float f) {
    unsigned u = __float_as_uint(f);
    return (u & 0x80000000u) ? ~u : (u | 0x80000000u);
}
__device__ inline float key_f32(unsigned k) {
    unsigned u = (k & 0x80000000u) ? (k ^ 0x80000000u) : ~k;
    return __uint_as_float(u);
}

__global__ void edge_max_kernel(const int* __restrict__ src, const int* __restrict__ dst,
                                int E, int E2, const float* __restrict__ a_s,
                                const float* __restrict__ a_d, unsigned* __restrict__ maxb)
{
    int t = blockIdx.x * blockDim.x + threadIdx.x;
    if (t >= E2 * HEADS) return;
    int e2 = t >> 2, h = t & 3, s, d;
    edge_sd(src, dst, E, e2, s, d);
    float l = a_s[s * HEADS + h] + a_d[d * HEADS + h];
    l = l >= 0.f ? l : 0.2f * l;
    atomicMax(&maxb[d * HEADS + h], f32_key(l));
}

__global__ void edge_exp_kernel(const int* __restrict__ src, const int* __restrict__ dst,
                                int E, int E2, const float* __restrict__ a_s,
                                const float* __restrict__ a_d, const unsigned* __restrict__ maxb,
                                float* __restrict__ exbuf, float* __restrict__ denom)
{
    int t = blockIdx.x * blockDim.x + threadIdx.x;
    if (t >= E2 * HEADS) return;
    int e2 = t >> 2, h = t & 3, s, d;
    edge_sd(src, dst, E, e2, s, d);
    float l = a_s[s * HEADS + h] + a_d[d * HEADS + h];
    l = l >= 0.f ? l : 0.2f * l;
    float m  = key_f32(maxb[d * HEADS + h]);
    float ex = __expf(l - m);
    exbuf[t] = ex;
    atomicAdd(&denom[d * HEADS + h], ex);
}

// gatout[dst,h,:] += alpha * xw[src,h,:] ; one wave per (edge,head)
__global__ __launch_bounds__(256) void edge_msg_kernel(
    const int* __restrict__ src, const int* __restrict__ dst, int E, int E2,
    const float* __restrict__ xw, const float* __restrict__ exbuf,
    const float* __restrict__ denom, float* __restrict__ gatout)
{
    int wave = threadIdx.x >> 5, lane = threadIdx.x & 31;
    int t = blockIdx.x * 8 + wave;
    if (t >= E2 * HEADS) return;
    int e2 = t >> 2, h = t & 3, s, d;
    edge_sd(src, dst, E, e2, s, d);
    float alpha = exbuf[t] / fmaxf(denom[d * HEADS + h], 1e-16f);
    float4 x = ((const float4*)(xw + (size_t)s * (HEADS * DH) + h * DH))[lane];
    float* o = gatout + (size_t)d * (HEADS * DH) + h * DH + lane * 4;
    atomicAdd(o + 0, alpha * x.x);
    atomicAdd(o + 1, alpha * x.y);
    atomicAdd(o + 2, alpha * x.z);
    atomicAdd(o + 3, alpha * x.w);
}

__global__ void head_mean_kernel(const float* __restrict__ gatout,
                                 const float* __restrict__ bg,
                                 float* __restrict__ out, long n128)
{
    long i = (long)blockIdx.x * blockDim.x + threadIdx.x;
    if (i >= n128) return;
    long n = i >> 7;
    int  c = (int)(i & 127);
    const float* p = gatout + n * (HEADS * DH) + c;
    out[i] = 0.25f * (p[0] + p[DH] + p[2 * DH] + p[3 * DH]) + bg[c];
}

// ------------------------------ final tiny layer ---------------------------
__global__ void mlp2_kernel(const float* __restrict__ hc1, const float* __restrict__ Wc2,
                            const float* __restrict__ bc2, float* __restrict__ out, int N)
{
    int n = blockIdx.x * blockDim.x + threadIdx.x;
    if (n >= N) return;
    float o0 = bc2[0], o1 = bc2[1];
    const float* r = hc1 + (size_t)n * 64;
#pragma unroll 8
    for (int k = 0; k < 64; ++k) {
        float v = r[k];
        o0 += v * Wc2[k * 2 + 0];
        o1 += v * Wc2[k * 2 + 1];
    }
    out[n * 2 + 0] = o0;
    out[n * 2 + 1] = o1;
}

// ---------------------------------------------------------------------------

extern "C" void kernel_launch(void* const* d_in, const int* in_sizes, int n_in,
                              void* d_out, int out_size, void* d_ws, size_t ws_size,
                              hipStream_t stream)
{
    const float* x       = (const float*)d_in[0];
    const int*   ei      = (const int*)  d_in[1];
    const float* Wl1     = (const float*)d_in[2];
    const float* bl1     = (const float*)d_in[3];
    const float* Wr1     = (const float*)d_in[4];
    const float* g1      = (const float*)d_in[5];
    const float* beta1   = (const float*)d_in[6];
    const float* Wg      = (const float*)d_in[7];
    const float* att_src = (const float*)d_in[8];
    const float* att_dst = (const float*)d_in[9];
    const float* bg      = (const float*)d_in[10];
    const float* g2      = (const float*)d_in[11];
    const float* beta2   = (const float*)d_in[12];
    const float* Wl3     = (const float*)d_in[13];
    const float* bl3     = (const float*)d_in[14];
    const float* Wr3     = (const float*)d_in[15];
    const float* Wc1     = (const float*)d_in[16];
    const float* bc1     = (const float*)d_in[17];
    const float* Wc2     = (const float*)d_in[18];
    const float* bc2     = (const float*)d_in[19];

    const int N  = in_sizes[0] / DH;
    const int E  = in_sizes[1] / 2;
    const int E2 = E + N;
    const int* src = ei;
    const int* dst = ei + E;

    // -------- workspace carve (256B aligned slabs) --------
    char* w = (char*)d_ws;
    auto carve = [&](size_t bytes) -> char* {
        char* p = w;
        w += (bytes + 255) & ~(size_t)255;
        return p;
    };
    float*    deg    = (float*)   carve((size_t)N * 4);
    float*    agg    = (float*)   carve((size_t)N * DH * 4);
    float*    h1     = (float*)   carve((size_t)N * DH * 4);
    float*    h2     = (float*)   carve((size_t)N * DH * 4);
    float*    h3     = (float*)   carve((size_t)N * DH * 4);
    float*    xw     = (float*)   carve((size_t)N * HEADS * DH * 4);
    float*    gatout = (float*)   carve((size_t)N * HEADS * DH * 4);
    float*    a_s    = (float*)   carve((size_t)N * HEADS * 4);
    float*    a_d    = (float*)   carve((size_t)N * HEADS * 4);
    unsigned* maxb   = (unsigned*)carve((size_t)N * HEADS * 4);
    float*    denom  = (float*)   carve((size_t)N * HEADS * 4);
    float*    exbuf  = (float*)   carve((size_t)E2 * HEADS * 4);
    float*    hc1    = (float*)   carve((size_t)N * 64 * 4);
    _Float16* pWl1   = (_Float16*)carve((size_t)128 * 128 * 2);
    _Float16* pWr1   = (_Float16*)carve((size_t)128 * 128 * 2);
    _Float16* pWg    = (_Float16*)carve((size_t)128 * 512 * 2);
    _Float16* pWl3   = (_Float16*)carve((size_t)128 * 128 * 2);
    _Float16* pWr3   = (_Float16*)carve((size_t)128 * 128 * 2);
    _Float16* pWc1   = (_Float16*)carve((size_t)128 * 64 * 2);

    const int mb = (N + 127) / 128;                 // GEMM row-tile blocks
    auto cdiv = [](long a, long b) { return (int)((a + b - 1) / b); };

    // -------- pack weights into WMMA B-fragment layout --------
    pack_w_kernel<<<cdiv(128 * 128, 256), 256, 0, stream>>>(Wl1, pWl1, 128, 128);
    pack_w_kernel<<<cdiv(128 * 128, 256), 256, 0, stream>>>(Wr1, pWr1, 128, 128);
    pack_w_kernel<<<cdiv(128 * 512, 256), 256, 0, stream>>>(Wg,  pWg,  128, 512);
    pack_w_kernel<<<cdiv(128 * 128, 256), 256, 0, stream>>>(Wl3, pWl3, 128, 128);
    pack_w_kernel<<<cdiv(128 * 128, 256), 256, 0, stream>>>(Wr3, pWr3, 128, 128);
    pack_w_kernel<<<cdiv(128 * 64, 256),  256, 0, stream>>>(Wc1, pWc1, 128, 64);

    // -------- SAGE1: agg = mean_j x_j ; h1 = agg@Wl1 + bl1 + x@Wr1 --------
    zero_kernel<<<cdiv(N, 256), 256, 0, stream>>>(deg, N);
    zero_kernel<<<cdiv((long)N * DH, 256), 256, 0, stream>>>(agg, (long)N * DH);
    deg_kernel<<<cdiv(E, 256), 256, 0, stream>>>(dst, deg, E);
    scatter_feat_kernel<<<cdiv(E, 8), 256, 0, stream>>>(x, src, dst, agg, E);
    div_deg_kernel<<<cdiv((long)N * DH, 256), 256, 0, stream>>>(agg, deg, (long)N * DH);
    gemm_kernel<8, 4, true><<<dim3(mb, 1), 256, 0, stream>>>(
        agg, (const v16h*)pWl1, x, (const v16h*)pWr1, bl1, h1, N, 128, 0);
    ln_act_kernel<<<cdiv(N, 8), 256, 0, stream>>>(h1, g1, beta1, N, 1);

    // -------- GAT: xw = h1@Wg ; segment softmax ; head mean --------
    gemm_kernel<8, 4, false><<<dim3(mb, 4), 256, 0, stream>>>(
        h1, (const v16h*)pWg, nullptr, nullptr, nullptr, xw, N, 512, 0);
    attn_coef_kernel<<<cdiv((long)N * HEADS, 8), 256, 0, stream>>>(xw, att_src, att_dst,
                                                                   a_s, a_d, N);
    zero_kernel<<<cdiv((long)N * HEADS, 256), 256, 0, stream>>>((float*)maxb, (long)N * HEADS);
    zero_kernel<<<cdiv((long)N * HEADS, 256), 256, 0, stream>>>(denom, (long)N * HEADS);
    zero_kernel<<<cdiv((long)N * HEADS * DH, 256), 256, 0, stream>>>(gatout, (long)N * HEADS * DH);
    edge_max_kernel<<<cdiv((long)E2 * HEADS, 256), 256, 0, stream>>>(src, dst, E, E2, a_s, a_d, maxb);
    edge_exp_kernel<<<cdiv((long)E2 * HEADS, 256), 256, 0, stream>>>(src, dst, E, E2, a_s, a_d,
                                                                     maxb, exbuf, denom);
    edge_msg_kernel<<<cdiv((long)E2 * HEADS, 8), 256, 0, stream>>>(src, dst, E, E2, xw,
                                                                   exbuf, denom, gatout);
    head_mean_kernel<<<cdiv((long)N * DH, 256), 256, 0, stream>>>(gatout, bg, h2, (long)N * DH);
    ln_act_kernel<<<cdiv(N, 8), 256, 0, stream>>>(h2, g2, beta2, N, 1);

    // -------- SAGE3: h3 = mean_j(h2)@Wl3 + bl3 + h2@Wr3 --------
    zero_kernel<<<cdiv((long)N * DH, 256), 256, 0, stream>>>(agg, (long)N * DH);
    scatter_feat_kernel<<<cdiv(E, 8), 256, 0, stream>>>(h2, src, dst, agg, E);
    div_deg_kernel<<<cdiv((long)N * DH, 256), 256, 0, stream>>>(agg, deg, (long)N * DH);
    gemm_kernel<8, 4, true><<<dim3(mb, 1), 256, 0, stream>>>(
        agg, (const v16h*)pWl3, h2, (const v16h*)pWr3, bl3, h3, N, 128, 0);

    // -------- classifier --------
    gemm_kernel<4, 4, false><<<dim3(mb, 1), 256, 0, stream>>>(
        h3, (const v16h*)pWc1, nullptr, nullptr, bc1, hc1, N, 64, 1);
    mlp2_kernel<<<cdiv(N, 256), 256, 0, stream>>>(hc1, Wc2, bc2, (float*)d_out, N);
}
